// VectorQuantizer_33191507264265
// MI455X (gfx1250) — matrix operand order repaired and checked by reading, not compiled, not run on hardware
//
#include <hip/hip_runtime.h>
#include <math.h>

typedef __attribute__((ext_vector_type(16))) __bf16 v16bf;
typedef __attribute__((ext_vector_type(8)))  float  v8f;

#define N_TOK   32768     // 32*32*32 tokens
#define E_DIM   256
#define N_E     1024
#define BETA    0.25f
#define Z_ELEMS 8388608   // 32*256*32*32

// d_out layout (float element offsets), outputs concatenated in return order:
// loss(1), z_q_st(8388608), perplexity(1), one_hot(33554432), indices(32768)
#define OUT_LOSS   0
#define OUT_ZQ     1
#define OUT_PERP   8388609
#define OUT_ONEHOT 8388610u
#define OUT_IDX    41943042u

// d_ws layout (byte offsets)
#define OFF_IDX    0u        // int[32768]
#define OFF_COUNTS 131072u   // int[1024]
#define OFF_PART   135168u   // float[32768]
#define OFF_EN2    266240u   // float[1024]
#define OFF_EHI    270336u   // bf16[1024*256] WMMA-B layout
#define OFF_ELO    794624u   // bf16[1024*256] WMMA-B layout

// ---------------------------------------------------------------------------
// Prep: split e_w (f32) into bf16 hi/lo, pre-swizzled into WMMA B-matrix
// layout: group g = (cc*8+kc)*32 + lane, each lane stores 16 contiguous K.
// 16384 threads.
__global__ void vq_prep_e(const float* __restrict__ e_w,
                          __bf16* __restrict__ ehi, __bf16* __restrict__ elo) {
  int g    = blockIdx.x * blockDim.x + threadIdx.x;   // 0..16383
  int lane = g & 31;
  int ckk  = g >> 5;                                  // cc*8 + kc
  int cc   = ckk >> 3, kc = ckk & 7;
  int j    = cc * 16 + (lane & 15);                   // code index (column)
  int kb   = kc * 32 + (lane >> 4) * 16;              // K base for this lane
  const float* src = e_w + (size_t)j * E_DIM + kb;
  v16bf vh, vl;
#pragma unroll
  for (int t = 0; t < 16; ++t) {
    float x   = src[t];
    __bf16 h  = (__bf16)x;
    float rem = x - (float)h;
    vh[t] = h;
    vl[t] = (__bf16)rem;
  }
  ((v16bf*)ehi)[g] = vh;
  ((v16bf*)elo)[g] = vl;
}

// ||e_j||^2, 1024 threads.
__global__ void vq_prep_en2(const float* __restrict__ e_w, float* __restrict__ en2) {
  int j = blockIdx.x * blockDim.x + threadIdx.x;
  float s = 0.f;
  const float* row = e_w + (size_t)j * E_DIM;
  for (int k = 0; k < E_DIM; ++k) { float x = row[k]; s = fmaf(x, x, s); }
  en2[j] = s;
}

// ---------------------------------------------------------------------------
// Async stage of one B group (2 col-chunks = 16 KB hi + 16 KB lo) into LDS
// buffer p. 64 slices of 512 B (32 lanes x b128), 8 per wave, via the CDNA5
// async Global->LDS data path (tracked by ASYNCcnt).
__device__ __forceinline__ void stage_group(const char* ehi_b, const char* elo_b,
                                            char* smem, int g, int p,
                                            int wid, int lane) {
#pragma unroll
  for (int i = 0; i < 8; ++i) {
    const int s    = wid * 8 + i;        // 0..63
    const int isLo = s >> 5;             // first 32 slices: hi, rest: lo
    const int s32  = s & 31;
    const char* src = (isLo ? elo_b : ehi_b) +
                      (size_t)g * 16384 + s32 * 512 + lane * 16;
    // LDS aperture: addr[31:0] of the generic pointer is the LDS byte offset.
    uint32_t dst = (uint32_t)(uintptr_t)(smem + p * 32768 + isLo * 16384 +
                                         s32 * 512 + lane * 16);
    asm volatile("global_load_async_to_lds_b128 %0, %1, off"
                 :: "v"(dst), "v"(src) : "memory");
  }
}

// ---------------------------------------------------------------------------
// Fused GEMM + argmin. Each wave owns 16 token rows (A-tile resident in
// VGPRs as bf16 hi/lo across full K=256). B (codebook) is staged into LDS
// cooperatively by the workgroup (double-buffered async copies) so each block
// reads the 1 MB codebook exactly once from L2. 3 bf16 WMMAs per K=32 step
// (hi*hi + hi*lo + lo*hi ~ fp32 accuracy); per-row running argmin; final
// 16-lane shfl_xor reduction (wave32).
__global__ void __launch_bounds__(256) vq_argmin(
    const float* __restrict__ z,
    const __bf16* __restrict__ ehi, const __bf16* __restrict__ elo,
    const float* __restrict__ en2, int* __restrict__ idx) {
  __shared__ __align__(32) char smem[65536];   // 2 x 32 KB double buffer

  const int lane = threadIdx.x & 31;
  const int wid  = threadIdx.x >> 5;
  const int rt   = blockIdx.x * 8 + wid;    // row tile id: rows rt*16..rt*16+15
  const int row  = lane & 15;               // A: lane (mod 16) holds row M=row
  const int half = lane >> 4;
  const int b    = rt >> 6;                 // batch (1024 tokens per batch)
  const int hw0  = (rt & 63) << 4;          // spatial base within batch
  // z[b, c, hw]: token n = b*1024+hw maps to addr b*262144 + c*1024 + hw
  const float* zb = z + (size_t)b * 262144 + hw0 + row;

  // A-matrix 16x32 bf16, this code's contiguous K order (same as B): lanes 0-15 K {0..15}; lanes 16-31 {16..31}
  v16bf Ahi[8], Alo[8];
#pragma unroll
  for (int kc = 0; kc < 8; ++kc) {
#pragma unroll
    for (int r2 = 0; r2 < 2; ++r2) {
      const int cbase = kc * 32 + half * 16 + r2 * 8;
#pragma unroll
      for (int t = 0; t < 8; ++t) {
        float x   = zb[(size_t)(cbase + t) * 1024];
        __bf16 h  = (__bf16)x;
        float rem = x - (float)h;
        Ahi[kc][r2 * 8 + t] = h;
        Alo[kc][r2 * 8 + t] = (__bf16)rem;
      }
    }
  }

  float minv[8]; int mini[8];
#pragma unroll
  for (int r = 0; r < 8; ++r) { minv[r] = 3.402823e38f; mini[r] = 0; }

  const char* ehi_b = (const char*)ehi;
  const char* elo_b = (const char*)elo;

  // prime the pipeline: group 0 -> buffer 0
  stage_group(ehi_b, elo_b, smem, 0, 0, wid, lane);

  for (int g = 0; g < 32; ++g) {           // 32 groups x 2 col-chunks
    const int p = g & 1;
    if (g + 1 < 32) {
      stage_group(ehi_b, elo_b, smem, g + 1, p ^ 1, wid, lane);
      // async loads complete in order: cnt<=8 means group g has landed
      asm volatile("s_wait_asynccnt 0x8" ::: "memory");
    } else {
      asm volatile("s_wait_asynccnt 0x0" ::: "memory");
    }
    __syncthreads();                        // group g visible to all waves

#pragma unroll
    for (int ccg = 0; ccg < 2; ++ccg) {
      const int cc = g * 2 + ccg;
      v8f acc = {0.f, 0.f, 0.f, 0.f, 0.f, 0.f, 0.f, 0.f};
#pragma unroll
      for (int kc = 0; kc < 8; ++kc) {
        const char* ph = smem + p * 32768 + ((ccg * 8 + kc) * 32 + lane) * 32;
        v16bf bh = *(const v16bf*)ph;
        v16bf bl = *(const v16bf*)(ph + 16384);
        acc = __builtin_amdgcn_wmma_f32_16x16x32_bf16(false, Ahi[kc], false, bh,
                                                      (short)0, acc, false, false);
        acc = __builtin_amdgcn_wmma_f32_16x16x32_bf16(false, Ahi[kc], false, bl,
                                                      (short)0, acc, false, false);
        acc = __builtin_amdgcn_wmma_f32_16x16x32_bf16(false, Alo[kc], false, bh,
                                                      (short)0, acc, false, false);
      }
      // C layout: VGPR r, lanes 0-15 -> (M=r, N=lane); lanes 16-31 -> (M=8+r, N=lane-16)
      const int j    = cc * 16 + row;       // column (code) this lane holds
      const float e2 = en2[j];              // d_rel = ||e||^2 - 2 z.e
#pragma unroll
      for (int r = 0; r < 8; ++r) {
        float s   = fmaf(-2.0f, acc[r], e2);
        bool take = s < minv[r];            // strict < keeps first index on ties
        minv[r] = take ? s : minv[r];
        mini[r] = take ? j : mini[r];
      }
    }
    __syncthreads();                        // done reading buf p before reuse
  }

  // reduce across the 16 lanes sharing each row (lane bits 0..3)
#pragma unroll
  for (int m = 1; m < 16; m <<= 1) {
#pragma unroll
    for (int r = 0; r < 8; ++r) {
      float ov = __shfl_xor(minv[r], m, 32);
      int   oi = __shfl_xor(mini[r], m, 32);
      bool take = (ov < minv[r]) || ((ov == minv[r]) && (oi < mini[r]));
      minv[r] = take ? ov : minv[r];
      mini[r] = take ? oi : mini[r];
    }
  }
  if (row == 0) {
    const int nbase = rt * 16 + half * 8;   // lanes 0-15: rows 0-7; 16-31: rows 8-15
#pragma unroll
    for (int r = 0; r < 8; ++r) idx[nbase + r] = mini[r];
  }
}

// ---------------------------------------------------------------------------
// Scatter: float indices, one-hot ones (region pre-zeroed), histogram counts.
__global__ void vq_scatter(const int* __restrict__ idx, int* __restrict__ counts,
                           float* __restrict__ onehot, float* __restrict__ out_idx) {
  int n = blockIdx.x * blockDim.x + threadIdx.x;     // 0..32767
  int id = idx[n];
  out_idx[n] = (float)id;
  atomicAdd(&counts[id], 1);
  onehot[(size_t)n * N_E + id] = 1.0f;
}

// z_q gather + per-block squared-error partials. One element per thread
// (coalesced b32 stores; e_w gathers are L2-resident, table = 1 MB).
__global__ void vq_zq_loss(const float* __restrict__ z, const float* __restrict__ e_w,
                           const int* __restrict__ idx, float* __restrict__ zq,
                           float* __restrict__ part) {
  __shared__ float sdata[256];
  size_t m  = (size_t)blockIdx.x * 256 + threadIdx.x; // b*262144 + c*1024 + hw
  int hw = (int)(m & 1023);
  size_t bc = m >> 10;
  int c = (int)(bc & 255);
  int b = (int)(bc >> 8);
  int n = b * 1024 + hw;
  int id = idx[n];
  float q  = e_w[(size_t)id * E_DIM + c];
  float zz = z[m];
  float d  = q - zz;
  zq[m] = q;                                 // straight-through value == z_q
  sdata[threadIdx.x] = d * d;
  __syncthreads();
  for (int s = 128; s > 0; s >>= 1) {
    if (threadIdx.x < s) sdata[threadIdx.x] += sdata[threadIdx.x + s];
    __syncthreads();
  }
  if (threadIdx.x == 0) part[blockIdx.x] = sdata[0];
}

// Finalize: loss and perplexity (double accumulation, deterministic).
__global__ void vq_finalize(const float* __restrict__ part, const int* __restrict__ counts,
                            float* __restrict__ out_loss, float* __restrict__ out_perp) {
  __shared__ double sd[256];
  int tid = threadIdx.x;
  double s = 0.0;
  for (int i = tid; i < 32768; i += 256) s += (double)part[i];
  sd[tid] = s; __syncthreads();
  for (int st = 128; st > 0; st >>= 1) { if (tid < st) sd[tid] += sd[tid + st]; __syncthreads(); }
  if (tid == 0) *out_loss = (float)((double)BETA * sd[0] / (double)Z_ELEMS);
  __syncthreads();
  double ps = 0.0;
  for (int j = tid; j < N_E; j += 256) {
    double p = (double)counts[j] / (double)N_TOK;
    ps += p * log(p + 1e-10);
  }
  sd[tid] = ps; __syncthreads();
  for (int st = 128; st > 0; st >>= 1) { if (tid < st) sd[tid] += sd[tid + st]; __syncthreads(); }
  if (tid == 0) *out_perp = (float)exp(-sd[0]);
}

// ---------------------------------------------------------------------------
extern "C" void kernel_launch(void* const* d_in, const int* in_sizes, int n_in,
                              void* d_out, int out_size, void* d_ws, size_t ws_size,
                              hipStream_t stream) {
  const float* z   = (const float*)d_in[0];   // (32,256,32,32)
  const float* e_w = (const float*)d_in[1];   // (1024,256)
  float* out = (float*)d_out;
  char*  ws  = (char*)d_ws;

  int*    idx    = (int*)(ws + OFF_IDX);
  int*    counts = (int*)(ws + OFF_COUNTS);
  float*  part   = (float*)(ws + OFF_PART);
  float*  en2    = (float*)(ws + OFF_EN2);
  __bf16* ehi    = (__bf16*)(ws + OFF_EHI);
  __bf16* elo    = (__bf16*)(ws + OFF_ELO);

  // zero one_hot output region and histogram (capture-safe memset nodes)
  hipMemsetAsync(out + OUT_ONEHOT, 0, (size_t)N_TOK * N_E * sizeof(float), stream);
  hipMemsetAsync(counts, 0, N_E * sizeof(int), stream);

  vq_prep_e  <<<64,    256, 0, stream>>>(e_w, ehi, elo);
  vq_prep_en2<<<4,     256, 0, stream>>>(e_w, en2);
  vq_argmin  <<<256,   256, 0, stream>>>(z, ehi, elo, en2, idx);
  vq_scatter <<<128,   256, 0, stream>>>(idx, counts, out + OUT_ONEHOT, out + OUT_IDX);
  vq_zq_loss <<<32768, 256, 0, stream>>>(z, e_w, idx, out + OUT_ZQ, part);
  vq_finalize<<<1,     256, 0, stream>>>(part, counts, out + OUT_LOSS, out + OUT_PERP);
}
